// SAGE_24575802868448
// MI455X (gfx1250) — compile-verified, matrix operand order checked by reading, not run on hardware
//
#include <hip/hip_runtime.h>

// ---------------------------------------------------------------------------
// GIN (5 layers) + MLP head for gfx1250 (MI455X).
// Matrix ops: v_wmma_f32_16x16x32_bf16 (bf16 A/B, fp32 accumulate).
// Aggregation: hardware float atomics (global_atomic_add_f32).
// ---------------------------------------------------------------------------

#define DEV __device__ __forceinline__

typedef __bf16 bf16;
typedef bf16  v16bf __attribute__((ext_vector_type(16)));
typedef bf16  v8bf  __attribute__((ext_vector_type(8)));
typedef float v8f   __attribute__((ext_vector_type(8)));

static constexpr int N_NODES = 50000;
static constexpr int N_EDGES = 800000;
static constexpr int F   = 128;   // in/hidden channels
static constexpr int C1  = 256;   // 2H
static constexpr int OUT = 41;
static constexpr int OUTP = 48;   // OUT padded to a multiple of 16
static constexpr int LNUM = 5;

DEV float gatom_add(float* p, float v) {
  return __hip_atomic_fetch_add(p, v, __ATOMIC_RELAXED, __HIP_MEMORY_SCOPE_AGENT);
}

// round-to-nearest-even float -> bf16, pure bit math (no fptrunc dependence)
DEV bf16 f2bf(float f) {
  unsigned u = __builtin_bit_cast(unsigned, f);
  u += 0x7fffu + ((u >> 16) & 1u);
  unsigned short s = (unsigned short)(u >> 16);
  return __builtin_bit_cast(bf16, s);
}

// --------------------------- elementwise helpers ---------------------------

__global__ void fill_zero(float* __restrict__ p, int n) {
  int i = blockIdx.x * blockDim.x + threadIdx.x;
  if (i < n) p[i] = 0.0f;
}

__global__ void conv_bf16(const float* __restrict__ in, bf16* __restrict__ out, int n) {
  int i = blockIdx.x * blockDim.x + threadIdx.x;
  if (i < n) out[i] = f2bf(in[i]);
}

// z = (1+eps[layer])*h + agg  ->  bf16
__global__ void make_z(const float* __restrict__ h, const float* __restrict__ agg,
                       const float* __restrict__ eps, int layer,
                       bf16* __restrict__ zb, int n) {
  int i = blockIdx.x * blockDim.x + threadIdx.x;
  if (i < n) {
    float e = 1.0f + eps[layer];
    zb[i] = f2bf(e * h[i] + agg[i]);
  }
}

// ----------------------------- edge aggregation ----------------------------
// one wave per edge, each lane handles 4 channels (float4 gather + 4 atomics)
__global__ void scatter_add(const float* __restrict__ hin,
                            const int* __restrict__ src, const int* __restrict__ dst,
                            float* __restrict__ agg) {
  int t = blockIdx.x * blockDim.x + threadIdx.x;
  int e = t >> 5;
  if (e >= N_EDGES) return;
  int lane = t & 31;
  const float4 v = *(const float4*)(hin + (size_t)src[e] * F + lane * 4);
  float* o = agg + (size_t)dst[e] * F + lane * 4;
  gatom_add(o + 0, v.x);
  gatom_add(o + 1, v.y);
  gatom_add(o + 2, v.z);
  gatom_add(o + 3, v.w);
}

// ------------------------------ weight packing -----------------------------
// Pack W [K x validCols] (row-major fp32) into the WMMA B-fragment lane layout
// for 16-col tiles: Bp[((ct*(K/32)+kt)*32 + lane)*16 + i]. Cols >= validCols = 0.
__global__ void pack_weights(const float* __restrict__ W, bf16* __restrict__ Bp,
                             int K, int ncols16, int validCols) {
  int idx = blockIdx.x * blockDim.x + threadIdx.x;
  int total = K * ncols16;
  if (idx >= total) return;
  int i    = idx & 15;
  int lane = (idx >> 4) & 31;
  int t    = idx >> 9;
  int ktc  = K >> 5;
  int kt   = t % ktc;
  int ct   = t / ktc;
  int v    = i >> 1;
  int k    = kt * 32 + ((lane >> 4) * 8) + ((v < 4) ? (2 * v) : (2 * v + 8)) + (i & 1);
  int col  = ct * 16 + (lane & 15);
  float w  = (col < validCols) ? W[(size_t)k * validCols + col] : 0.0f;
  Bp[idx]  = f2bf(w);
}

// --------------------------------- GEMM ------------------------------------
// C[nrows x ldc] = A[nrows x K](bf16) * Bpacked(K x gridDim.y*16) + bias
// Block = 8 waves; each wave owns one 16x16 tile (block covers 128 rows).
__global__ void __launch_bounds__(256)
gemm_bf16_wmma(const bf16* __restrict__ A, const bf16* __restrict__ Bp,
               const float* __restrict__ bias, float* __restrict__ C,
               int nrows, int K, int ldc, int validCols) {
  const int lane = threadIdx.x & 31;
  const int wave = threadIdx.x >> 5;
  const int rowBase = blockIdx.x * 128 + wave * 16;
  const int m  = lane & 15;
  const int hi = lane >> 4;
  const int ktc = K >> 5;

  int arow = rowBase + m;
  if (arow >= nrows) arow = nrows - 1;  // clamped rows pollute only unstored D rows
  const bf16* __restrict__ Aw = A + (size_t)arow * K + hi * 8;
  const bf16* __restrict__ Bw = Bp + ((size_t)blockIdx.y * ktc * 32 + lane) * 16;

  v8f acc = {};
  for (int kt = 0; kt < ktc; ++kt) {
    const bf16* ap = Aw + kt * 32;
    v8bf a0 = *(const v8bf*)(ap);        // K = k0+khalf .. +7
    v8bf a1 = *(const v8bf*)(ap + 16);   // K = k0+khalf+16 .. +23
    const bf16* bp = Bw + (size_t)kt * 512;
    v8bf b0 = *(const v8bf*)(bp);
    v8bf b1 = *(const v8bf*)(bp + 8);
    v16bf a = __builtin_shufflevector(a0, a1, 0,1,2,3,4,5,6,7,8,9,10,11,12,13,14,15);
    v16bf b = __builtin_shufflevector(b0, b1, 0,1,2,3,4,5,6,7,8,9,10,11,12,13,14,15);
    acc = __builtin_amdgcn_wmma_f32_16x16x32_bf16(false, a, false, b,
                                                  (short)0, acc, false, false);
  }

  const int col = blockIdx.y * 16 + m;
  if (col < validCols) {
    const float bs = bias[col];
    const int orow0 = rowBase + hi * 8;
#pragma unroll
    for (int r = 0; r < 8; ++r) {
      int orow = orow0 + r;
      if (orow < nrows) C[(size_t)orow * ldc + col] = acc[r] + bs;
    }
  }
}

// ------------------------------ batch-norm ---------------------------------
// Per-thread column-resident accumulation: stride is a multiple of C, so each
// thread always touches the same column; then a few global f32 atomics.
__global__ void bn_stats(const float* __restrict__ Y, float* __restrict__ sums,
                         float* __restrict__ sqs, int total, int C) {
  int idx0 = blockIdx.x * blockDim.x + threadIdx.x;
  int stride = gridDim.x * blockDim.x;   // 256*256: multiple of 128 and 256
  float s = 0.0f, q = 0.0f;
  for (int i = idx0; i < total; i += stride) {
    float v = Y[i];
    s += v;
    q += v * v;
  }
  int col = idx0 % C;
  gatom_add(&sums[col], s);
  gatom_add(&sqs[col], q);
}

__global__ void bn_finalize(const float* __restrict__ sums, const float* __restrict__ sqs,
                            float* __restrict__ mean, float* __restrict__ rstd,
                            int C, float invN) {
  int c = blockIdx.x * blockDim.x + threadIdx.x;
  if (c < C) {
    float m = sums[c] * invN;
    float v = sqs[c] * invN - m * m;     // biased variance (matches jnp.var)
    mean[c] = m;
    rstd[c] = rsqrtf(v + 1e-5f);
  }
}

__global__ void bn_relu_bf16(const float* __restrict__ y, const float* __restrict__ g,
                             const float* __restrict__ beta, const float* __restrict__ mean,
                             const float* __restrict__ rstd, bf16* __restrict__ out,
                             int total, int C) {
  int i = blockIdx.x * blockDim.x + threadIdx.x;
  if (i < total) {
    int c = i % C;
    float v = g[c] * (y[i] - mean[c]) * rstd[c] + beta[c];
    out[i] = f2bf(fmaxf(v, 0.0f));
  }
}

__global__ void bn_relu_f32(const float* __restrict__ y, const float* __restrict__ g,
                            const float* __restrict__ beta, const float* __restrict__ mean,
                            const float* __restrict__ rstd, float* __restrict__ out,
                            int total, int C) {
  int i = blockIdx.x * blockDim.x + threadIdx.x;
  if (i < total) {
    int c = i % C;
    float v = g[c] * (y[i] - mean[c]) * rstd[c] + beta[c];
    out[i] = fmaxf(v, 0.0f);
  }
}

// ------------------------------ log-softmax --------------------------------
__global__ void log_softmax41(const float* __restrict__ logits, float* __restrict__ out,
                              int nrows) {
  int row = blockIdx.x * 8 + (threadIdx.x >> 5);
  if (row >= nrows) return;
  int lane = threadIdx.x & 31;
  const float* p = logits + (size_t)row * OUTP;
  float a = (lane < OUT)      ? p[lane]      : -3.4e38f;
  float b = (lane + 32 < OUT) ? p[lane + 32] : -3.4e38f;
  float mx = fmaxf(a, b);
  for (int s = 16; s; s >>= 1) mx = fmaxf(mx, __shfl_xor(mx, s, 32));
  float e = ((lane < OUT) ? __expf(a - mx) : 0.0f) +
            ((lane + 32 < OUT) ? __expf(b - mx) : 0.0f);
  for (int s = 16; s; s >>= 1) e += __shfl_xor(e, s, 32);
  float lse = mx + __logf(e);
  if (lane < OUT)      out[(size_t)row * OUT + lane]      = a - lse;
  if (lane + 32 < OUT) out[(size_t)row * OUT + lane + 32] = b - lse;
}

// ------------------------------- launcher ----------------------------------
extern "C" void kernel_launch(void* const* d_in, const int* in_sizes, int n_in,
                              void* d_out, int out_size, void* d_ws, size_t ws_size,
                              hipStream_t stream) {
  const float* x      = (const float*)d_in[0];
  const int*   ei     = (const int*)  d_in[1];
  const float* W1     = (const float*)d_in[2];
  const float* b1     = (const float*)d_in[3];
  const float* g1     = (const float*)d_in[4];
  const float* beta1  = (const float*)d_in[5];
  const float* W2     = (const float*)d_in[6];
  const float* b2     = (const float*)d_in[7];
  const float* epsp   = (const float*)d_in[8];
  const float* g2     = (const float*)d_in[9];
  const float* beta2  = (const float*)d_in[10];
  const float* lin1_w = (const float*)d_in[11];
  const float* lin1_b = (const float*)d_in[12];
  const float* bn1_g  = (const float*)d_in[13];
  const float* bn1_b  = (const float*)d_in[14];
  const float* lin2_w = (const float*)d_in[15];
  const float* lin2_b = (const float*)d_in[16];
  float* out = (float*)d_out;

  char* base = (char*)d_ws;
  auto alloc = [&](size_t bytes) -> char* {
    char* p = base;
    base += (bytes + 255) & ~(size_t)255;
    return p;
  };
  float* agg = (float*)alloc((size_t)N_NODES * F  * 4);  // also final logits [N,48]
  float* h   = (float*)alloc((size_t)N_NODES * F  * 4);
  float* y1  = (float*)alloc((size_t)N_NODES * C1 * 4);  // also y3 [N,128]
  bf16*  zb  = (bf16*) alloc((size_t)N_NODES * F  * 2);  // also hb
  bf16*  z2b = (bf16*) alloc((size_t)N_NODES * C1 * 2);  // also z3b
  bf16*  W1p = (bf16*) alloc((size_t)LNUM * F  * C1 * 2);
  bf16*  W2p = (bf16*) alloc((size_t)LNUM * C1 * F  * 2);
  bf16*  l1p = (bf16*) alloc((size_t)F * F * 2);
  bf16*  l2p = (bf16*) alloc((size_t)F * OUTP * 2);
  float* sums  = (float*)alloc(4 * 256 * 4);
  float* sqs   = sums + 256;
  float* meanv = sums + 512;
  float* rstdv = sums + 768;

  const int NF  = N_NODES * F;    // 6.4M
  const int NC1 = N_NODES * C1;   // 12.8M
  const int gx  = (N_NODES + 127) / 128;
  const float invN = 1.0f / (float)N_NODES;

  // ---- pack weights to bf16 fragment layout (small, once per launch) ----
  for (int i = 0; i < LNUM; ++i)
    pack_weights<<<(F * C1 + 255) / 256, 256, 0, stream>>>(
        W1 + (size_t)i * F * C1, W1p + (size_t)i * F * C1, F, C1, C1);
  for (int i = 0; i < LNUM; ++i)
    pack_weights<<<(C1 * F + 255) / 256, 256, 0, stream>>>(
        W2 + (size_t)i * C1 * F, W2p + (size_t)i * C1 * F, C1, F, F);
  pack_weights<<<(F * F + 255) / 256, 256, 0, stream>>>(lin1_w, l1p, F, F, F);
  pack_weights<<<(F * OUTP + 255) / 256, 256, 0, stream>>>(lin2_w, l2p, F, OUTP, OUT);

  const float* hin = x;
  for (int i = 0; i < LNUM; ++i) {
    // sum-aggregation of neighbors
    fill_zero<<<(NF + 255) / 256, 256, 0, stream>>>(agg, NF);
    scatter_add<<<(N_EDGES * 32) / 256, 256, 0, stream>>>(hin, ei, ei + N_EDGES, agg);
    // z = (1+eps)*h + agg -> bf16
    make_z<<<(NF + 255) / 256, 256, 0, stream>>>(hin, agg, epsp, i, zb, NF);
    // y1 = z @ W1 + b1   [N,256]
    gemm_bf16_wmma<<<dim3(gx, C1 / 16), 256, 0, stream>>>(
        zb, W1p + (size_t)i * F * C1, b1 + i * C1, y1, N_NODES, F, C1, C1);
    // BN + ReLU -> bf16
    fill_zero<<<2, 256, 0, stream>>>(sums, 512);
    bn_stats<<<256, 256, 0, stream>>>(y1, sums, sqs, NC1, C1);
    bn_finalize<<<1, 256, 0, stream>>>(sums, sqs, meanv, rstdv, C1, invN);
    bn_relu_bf16<<<(NC1 + 255) / 256, 256, 0, stream>>>(
        y1, g1 + i * C1, beta1 + i * C1, meanv, rstdv, z2b, NC1, C1);
    // h = z2 @ W2 + b2   [N,128]
    gemm_bf16_wmma<<<dim3(gx, F / 16), 256, 0, stream>>>(
        z2b, W2p + (size_t)i * C1 * F, b2 + i * F, h, N_NODES, C1, F, F);
    if (i != LNUM - 1) {
      fill_zero<<<2, 256, 0, stream>>>(sums, 512);
      bn_stats<<<256, 256, 0, stream>>>(h, sums, sqs, NF, F);
      bn_finalize<<<1, 256, 0, stream>>>(sums, sqs, meanv, rstdv, F, invN);
      bn_relu_f32<<<(NF + 255) / 256, 256, 0, stream>>>(
          h, g2 + i * F, beta2 + i * F, meanv, rstdv, h, NF, F);
    }
    hin = h;
  }

  // ---- MLP head ----
  conv_bf16<<<(NF + 255) / 256, 256, 0, stream>>>(h, zb, NF);
  gemm_bf16_wmma<<<dim3(gx, F / 16), 256, 0, stream>>>(
      zb, l1p, lin1_b, y1, N_NODES, F, F, F);
  fill_zero<<<2, 256, 0, stream>>>(sums, 512);
  bn_stats<<<256, 256, 0, stream>>>(y1, sums, sqs, NF, F);
  bn_finalize<<<1, 256, 0, stream>>>(sums, sqs, meanv, rstdv, F, invN);
  bn_relu_bf16<<<(NF + 255) / 256, 256, 0, stream>>>(
      y1, bn1_g, bn1_b, meanv, rstdv, z2b, NF, F);
  // logits [N,48] (cols 41..47 unused), bias applied for cols < 41
  gemm_bf16_wmma<<<dim3(gx, OUTP / 16), 256, 0, stream>>>(
      z2b, l2p, lin2_b, agg, N_NODES, F, OUTP, OUT);
  log_softmax41<<<(N_NODES + 7) / 8, 256, 0, stream>>>(agg, out, N_NODES);
}
